// SimplifiedMultiHeadAttention_33234456936785
// MI455X (gfx1250) — compile-verified, matrix operand order checked by reading, not run on hardware
//
#include <hip/hip_runtime.h>

// Problem constants (from the reference)
#define MODEL_DIM 768
#define N_HEAD    12
#define HEAD_DIM  64
#define BATCH     2
#define SEQ       2048
#define M_TOK     (BATCH * SEQ)      // 4096 tokens total
#define QTILES    (SEQ / 16)         // 128
#define NTILES_D  (MODEL_DIM / 16)   // 48

// LDS layout for the fused attention kernel (per workgroup):
//  scores: [16][SCORE_STRIDE] f32 ; stride 2068: 2068 % 64 = 20, gcd(20,64)=4
//          -> 16 lanes x 4-dword runs tile the 64 banks conflict-free
//  partials: [8 waves][16][68] f32 for deterministic cross-wave reduction
#define SCORE_STRIDE 2068
#define PART_STRIDE  (16 * 68)
#define FUSED_SMEM_BYTES ((16 * SCORE_STRIDE + 8 * PART_STRIDE) * 4)  // 167,168 B

typedef __attribute__((ext_vector_type(16))) _Float16 v16h;
typedef __attribute__((ext_vector_type(8)))  float    v8f;

// ---------------------------------------------------------------------------
// WMMA fragment loaders (CDNA5 ISA 7.12.2, wave32, 16-bit 16x32 A / 32x16 B)
// A: lane l<16 -> row l, halves {K..K+7} then {K+16..K+23}; lane l>=16 -> row
// l-16, halves {K+8..K+15} then {K+24..K+31}. B uses the same striping with
// the lane selecting the output column (NT GEMM: contiguous-K rows).
// ---------------------------------------------------------------------------
__device__ __forceinline__ v16h frag_rowk_f32(const float* base, int stride,
                                              int row0, int kbase, int lane) {
  const int r  = row0 + (lane & 15);
  const int kh = (lane >> 4) << 3;   // 0 or 8
  const float* p = base + (size_t)r * stride + kbase + kh;
  v16h f;
#pragma unroll
  for (int i = 0; i < 8; ++i) f[i] = (_Float16)p[i];
#pragma unroll
  for (int i = 0; i < 8; ++i) f[8 + i] = (_Float16)p[16 + i];
  return f;
}

__device__ __forceinline__ v16h frag_rowk_f16(const _Float16* __restrict__ base,
                                              int stride, int row0, int kbase,
                                              int lane) {
  const int r  = row0 + (lane & 15);
  const int kh = (lane >> 4) << 3;
  const _Float16* p = base + (size_t)r * stride + kbase + kh;
  v16h f;
#pragma unroll
  for (int i = 0; i < 8; ++i) f[i] = p[i];
#pragma unroll
  for (int i = 0; i < 8; ++i) f[8 + i] = p[16 + i];
  return f;
}

#define WMMA_F16F32(c, a, b)                                                   \
  (c) = __builtin_amdgcn_wmma_f32_16x16x32_f16(false, (a), false, (b),         \
                                               (short)0, (c), false, false)

// ---------------------------------------------------------------------------
// Kernel 1: Y16 = f16( X @ W^T + bias ), X [M_TOK,768] f32, W [768,768] f32.
// One wave computes a 16x64 strip (1x4 N-tiling). TRANSPOSE_OUT=true writes
// Yt[b, n, s] (stride SEQ) so the context GEMM gets contiguous-K B loads.
// ---------------------------------------------------------------------------
template <bool TRANSPOSE_OUT>
__global__ __launch_bounds__(256) void mha_proj_gemm(
    const float* __restrict__ X, const float* __restrict__ W,
    const float* __restrict__ bias, _Float16* __restrict__ Y) {
  const int lane = threadIdx.x & 31;
  const int wave = threadIdx.x >> 5;
  const int tile = blockIdx.x * 8 + wave;      // [mt][ntg], ntg in 0..11
  const int mt  = tile / (NTILES_D / 4);
  const int nt0 = (tile % (NTILES_D / 4)) * 4;

  v8f acc[4] = {};
  for (int kb = 0; kb < MODEL_DIM; kb += 32) {
    __builtin_prefetch(X + (size_t)(mt * 16 + (lane & 15)) * MODEL_DIM + kb + 64, 0, 1);
    v16h a = frag_rowk_f32(X, MODEL_DIM, mt * 16, kb, lane);
#pragma unroll
    for (int j = 0; j < 4; ++j) {
      v16h b = frag_rowk_f32(W, MODEL_DIM, (nt0 + j) * 16, kb, lane);
      WMMA_F16F32(acc[j], a, b);
    }
  }

  const int rbase = mt * 16 + ((lane >> 4) << 3);
#pragma unroll
  for (int j = 0; j < 4; ++j) {
    const int col  = (nt0 + j) * 16 + (lane & 15);
    const float bv = bias[col];
#pragma unroll
    for (int i = 0; i < 8; ++i) {
      const int r = rbase + i;
      const _Float16 val = (_Float16)(acc[j][i] + bv);
      if (TRANSPOSE_OUT) {
        const int b2 = r >> 11;            // token row -> (batch, seq)
        const int s  = r & (SEQ - 1);
        Y[((size_t)b2 * MODEL_DIM + col) * SEQ + s] = val;
      } else {
        Y[(size_t)r * MODEL_DIM + col] = val;
      }
    }
  }
}

// ---------------------------------------------------------------------------
// Kernel 2 (fused): scores -> softmax -> context for one (b, h, 16-row qtile).
// Scores live in LDS; attn_weights f32 written exactly once; context partials
// reduced deterministically across waves (fixed order, no float atomics).
// ---------------------------------------------------------------------------
__global__ __launch_bounds__(256) void mha_fused_attn(
    const _Float16* __restrict__ Q, const _Float16* __restrict__ K,
    const _Float16* __restrict__ Vt, float* __restrict__ attn,
    _Float16* __restrict__ Ctx) {
  extern __shared__ char smem_raw[];
  float* S    = (float*)smem_raw;                // [16][SCORE_STRIDE]
  float* Part = S + 16 * SCORE_STRIDE;           // [8][16][68]

  const int lane = threadIdx.x & 31;
  const int wave = threadIdx.x >> 5;
  int tile = blockIdx.x;                         // [b][h][qt]
  const int qt = tile & (QTILES - 1); tile >>= 7;
  const int h  = tile % N_HEAD;
  const int b  = tile / N_HEAD;

  const _Float16* Qb = Q + (size_t)b * SEQ * MODEL_DIM;
  const _Float16* Kb = K + (size_t)b * SEQ * MODEL_DIM;
  const int rbase = (lane >> 4) << 3;

  // ---- Phase 1: 16x2048 scaled scores into LDS ----------------------------
  const v16h a0 = frag_rowk_f16(Qb, MODEL_DIM, qt * 16, h * HEAD_DIM + 0,  lane);
  const v16h a1 = frag_rowk_f16(Qb, MODEL_DIM, qt * 16, h * HEAD_DIM + 32, lane);
  for (int p = 0; p < 4; ++p) {
    const int kt0 = (p * 8 + wave) * 4;
    v8f acc[4] = {};
#pragma unroll
    for (int j = 0; j < 4; ++j) {
      v16h b0 = frag_rowk_f16(Kb, MODEL_DIM, (kt0 + j) * 16, h * HEAD_DIM + 0, lane);
      WMMA_F16F32(acc[j], a0, b0);
      v16h b1 = frag_rowk_f16(Kb, MODEL_DIM, (kt0 + j) * 16, h * HEAD_DIM + 32, lane);
      WMMA_F16F32(acc[j], a1, b1);
    }
#pragma unroll
    for (int j = 0; j < 4; ++j) {
      const int col = (kt0 + j) * 16 + (lane & 15);
#pragma unroll
      for (int i = 0; i < 8; ++i)
        S[(rbase + i) * SCORE_STRIDE + col] = acc[j][i] * 0.125f;  // 1/sqrt(64)
    }
  }
  __syncthreads();

  // ---- Phase 2: row softmax (2 rows per wave, wave32 shuffles) ------------
  float* attn_bh = attn + (size_t)(b * N_HEAD + h) * SEQ * SEQ +
                   (size_t)qt * 16 * SEQ;
#pragma unroll
  for (int rr = 0; rr < 2; ++rr) {
    const int row = wave * 2 + rr;
    float* r = S + row * SCORE_STRIDE;
    float m = -3.402823466e38f;
    for (int c = lane; c < SEQ; c += 32) m = fmaxf(m, r[c]);
#pragma unroll
    for (int o = 16; o > 0; o >>= 1) m = fmaxf(m, __shfl_xor(m, o, 32));
    float sum = 0.f;
    for (int c = lane; c < SEQ; c += 32) {
      const float e = __expf(r[c] - m);
      r[c] = e;
      sum += e;
    }
#pragma unroll
    for (int o = 16; o > 0; o >>= 1) sum += __shfl_xor(sum, o, 32);
    const float inv = 1.0f / sum;
    float* g = attn_bh + (size_t)row * SEQ;
    for (int c = lane; c < SEQ; c += 32) {
      const float pval = r[c] * inv;
      r[c] = pval;                 // keep normalized P in LDS for phase 3
      g[c] = pval;                 // the one mandatory HBM write of attn
    }
  }
  __syncthreads();

  // ---- Phase 3: context partials, K-split across the 8 waves --------------
  const _Float16* Vtb = Vt + (size_t)b * MODEL_DIM * SEQ;
  v8f cacc[4] = {};
  const int kstart = wave * (SEQ / 8);
  for (int kb = kstart; kb < kstart + SEQ / 8; kb += 32) {
    v16h a = frag_rowk_f32(S, SCORE_STRIDE, 0, kb, lane);   // ds_load + cvt
#pragma unroll
    for (int j = 0; j < 4; ++j) {
      v16h bf = frag_rowk_f16(Vtb, SEQ, h * HEAD_DIM + j * 16, kb, lane);
      WMMA_F16F32(cacc[j], a, bf);
    }
  }
  float* mypart = Part + wave * PART_STRIDE;
#pragma unroll
  for (int j = 0; j < 4; ++j)
#pragma unroll
    for (int i = 0; i < 8; ++i)
      mypart[(rbase + i) * 68 + j * 16 + (lane & 15)] = cacc[j][i];
  __syncthreads();

  // ---- Phase 4: deterministic cross-wave reduce + f16 context store -------
  for (int idx = threadIdx.x; idx < 16 * HEAD_DIM; idx += 256) {
    const int row = idx >> 6;
    const int d   = idx & 63;
    float s = 0.f;
#pragma unroll
    for (int w = 0; w < 8; ++w) s += Part[w * PART_STRIDE + row * 68 + d];
    Ctx[((size_t)b * SEQ + qt * 16 + row) * MODEL_DIM + h * HEAD_DIM + d] =
        (_Float16)s;
  }
}

// ---------------------------------------------------------------------------
// Kernel 3: out = Ctx @ Wo^T + bo (f32 output), 1x4 N-tiling
// ---------------------------------------------------------------------------
__global__ __launch_bounds__(256) void mha_out_gemm(
    const _Float16* __restrict__ Ctx, const float* __restrict__ W,
    const float* __restrict__ bias, float* __restrict__ Y) {
  const int lane = threadIdx.x & 31;
  const int wave = threadIdx.x >> 5;
  const int tile = blockIdx.x * 8 + wave;
  const int mt  = tile / (NTILES_D / 4);
  const int nt0 = (tile % (NTILES_D / 4)) * 4;

  v8f acc[4] = {};
  for (int kb = 0; kb < MODEL_DIM; kb += 32) {
    v16h a = frag_rowk_f16(Ctx, MODEL_DIM, mt * 16, kb, lane);
#pragma unroll
    for (int j = 0; j < 4; ++j) {
      v16h b = frag_rowk_f32(W, MODEL_DIM, (nt0 + j) * 16, kb, lane);
      WMMA_F16F32(acc[j], a, b);
    }
  }

  const int rbase = mt * 16 + ((lane >> 4) << 3);
#pragma unroll
  for (int j = 0; j < 4; ++j) {
    const int col  = (nt0 + j) * 16 + (lane & 15);
    const float bv = bias[col];
#pragma unroll
    for (int i = 0; i < 8; ++i)
      Y[(size_t)(rbase + i) * MODEL_DIM + col] = acc[j][i] + bv;
  }
}

// ---------------------------------------------------------------------------
extern "C" void kernel_launch(void* const* d_in, const int* in_sizes, int n_in,
                              void* d_out, int out_size, void* d_ws,
                              size_t ws_size, hipStream_t stream) {
  (void)in_sizes; (void)n_in; (void)out_size; (void)ws_size;

  const float* q    = (const float*)d_in[0];
  const float* k    = (const float*)d_in[1];
  const float* v    = (const float*)d_in[2];
  const float* wq_w = (const float*)d_in[3];
  const float* wq_b = (const float*)d_in[4];
  const float* wk_w = (const float*)d_in[5];
  const float* wk_b = (const float*)d_in[6];
  const float* wv_w = (const float*)d_in[7];
  const float* wv_b = (const float*)d_in[8];
  const float* wo_w = (const float*)d_in[9];
  const float* wo_b = (const float*)d_in[10];

  float* out  = (float*)d_out;                       // [B,S,768]
  float* attn = out + (size_t)M_TOK * MODEL_DIM;     // [B,H,S,S]

  const size_t PROJ = (size_t)M_TOK * MODEL_DIM;     // 3,145,728 elems
  _Float16* q16  = (_Float16*)d_ws;                  // 4 x 6.29 MB = 25.2 MB
  _Float16* k16  = q16 + PROJ;
  _Float16* v16t = k16 + PROJ;                       // transposed [B,768,S]
  _Float16* c16  = v16t + PROJ;

  // Allow >64KB dynamic LDS for the fused kernel (320KB/WG is the HW limit).
  (void)hipFuncSetAttribute(reinterpret_cast<const void*>(mha_fused_attn),
                            hipFuncAttributeMaxDynamicSharedMemorySize,
                            FUSED_SMEM_BYTES);

  dim3 blk(256);
  // 256 mtiles * 12 n-groups = 3072 waves -> 384 blocks
  mha_proj_gemm<false><<<384, blk, 0, stream>>>(q, wq_w, wq_b, q16);
  mha_proj_gemm<false><<<384, blk, 0, stream>>>(k, wk_w, wk_b, k16);
  mha_proj_gemm<true ><<<384, blk, 0, stream>>>(v, wv_w, wv_b, v16t);

  // one workgroup per (b,h,qtile): 2*12*128 = 3072 blocks
  mha_fused_attn<<<3072, blk, FUSED_SMEM_BYTES, stream>>>(q16, k16, v16t,
                                                          attn, c16);

  // 3072 waves -> 384 blocks
  mha_out_gemm<<<384, blk, 0, stream>>>(c16, wo_w, wo_b, out);
}
